// myModel_82746839924790
// MI455X (gfx1250) — matrix operand order repaired and checked by reading, not compiled, not run on hardware
//
#include <hip/hip_runtime.h>
#include <hip/hip_bf16.h>
#include <math.h>

// ---------------------------------------------------------------------------
// RouteNet-style GNN on MI455X (gfx1250), wave32 + WMMA bf16.
//  - All GEMMs: v_wmma_f32_16x16x32_bf16, f32 accumulation.
//  - Readout GRU fused into the T-step loop (no [T,E,D] seq buffer).
//  - Packed B weights staged into LDS per block via the Tensor Data Mover
//    (tensor_load_to_lds + s_wait_tensorcnt), fallback cooperative copy.
//  - B fragments then come from ds_load_b128, keeping VMEM free for the
//    random gathers and f32 atomic scatters.
// ---------------------------------------------------------------------------

typedef __attribute__((ext_vector_type(16))) __bf16 v16bf;
typedef __attribute__((ext_vector_type(8)))  float  v8f;
typedef __attribute__((ext_vector_type(4)))  unsigned v4u;
typedef __attribute__((ext_vector_type(8)))  int      v8i;
typedef __attribute__((ext_vector_type(4)))  int      v4i;

namespace {
constexpr int kT = 8;        // message passing steps
constexpr int kD = 64;       // link state dim
constexpr int kR = 128;      // readout units
constexpr int kE = 100000;   // edges
constexpr int kM = 400000;   // messages
constexpr int kG = 1000;     // graphs

constexpr float kSeluScale      = 1.0507009873554805f;
constexpr float kSeluScaleAlpha = 1.7580993408473766f; // scale * alpha
}

#if defined(__has_builtin)
#  if __has_builtin(__builtin_amdgcn_tensor_load_to_lds) && (__clang_major__ < 23)
#    define USE_TDM 1
#  endif
#endif
#ifndef USE_TDM
#  define USE_TDM 0
#endif

__device__ __forceinline__ float sigmoidf_(float x) {
  return 1.0f / (1.0f + __expf(-x));
}

__device__ __forceinline__ v8f wmma_bf16(v16bf a, v16bf b, v8f c) {
  // (neg_a, A, neg_b, B, c_mod, C, reuse_a, reuse_b)
  return __builtin_amdgcn_wmma_f32_16x16x32_bf16(false, a, false, b,
                                                 (short)0, c, false, false);
}

// A fragment (16x32 bf16) for one lane from an f32 row-major source.
//   lanes 0-15  : K = {k0+0..7,  k0+16..23}
//   lanes 16-31 : K = {k0+8..15, k0+24..31}
__device__ __forceinline__ v16bf a_frag_f32(const float* __restrict__ rowptr,
                                            int half) {
  const float* p = rowptr + half * 8;
  v16bf a;
#pragma unroll
  for (int i = 0; i < 8; ++i) a[i] = (__bf16)p[i];
#pragma unroll
  for (int i = 0; i < 8; ++i) a[8 + i] = (__bf16)p[16 + i];
  return a;
}

// B fragment from the pre-packed tile layout: tile base + lane*16, 32B aligned.
// Works on both LDS (ds_load_b128) and global (global_load_b128) sources.
__device__ __forceinline__ v16bf b_frag(const __bf16* tile, int lane) {
  return *(const v16bf*)(tile + lane * 16);
}

// Stage an h x w_dw (dwords) tile, row stride stride_dw dwords, from global
// into contiguous LDS. TDM path: one descriptor, issued by wave 0.
__device__ __forceinline__ void stage_tiles(__bf16* dst_lds, const __bf16* src,
                                            unsigned w_dw, unsigned h,
                                            unsigned stride_dw) {
#if USE_TDM
  if ((threadIdx.x >> 5) == 0) {
    const unsigned lds = (unsigned)(unsigned long long)dst_lds;  // addr[31:0]
    const unsigned long long ga = (unsigned long long)src;
    v4u g0;
    g0[0] = 1u;                                   // count=1, user descriptor
    g0[1] = lds;                                  // lds_addr
    g0[2] = (unsigned)ga;                         // global_addr[31:0]
    g0[3] = (unsigned)((ga >> 32) & 0x01FFFFFFu)  // global_addr[56:32]
            | (2u << 30);                         // type = 2 ("image")
    v8i g1;
    g1[0] = (int)(2u << 16);                              // data_size = 4B
    g1[1] = (int)((w_dw & 0xFFFFu) << 16);                // tensor_dim0 lo
    g1[2] = (int)((w_dw >> 16) | ((h & 0xFFFFu) << 16));  // td0 hi | td1 lo
    g1[3] = (int)((h >> 16) | (w_dw << 16));              // td1 hi | tile_dim0
    g1[4] = (int)h;                                       // tile_dim1 (dim2=0)
    g1[5] = (int)stride_dw;                               // tensor_dim0_stride
    g1[6] = 0;
    g1[7] = 0;
    v4i z = {0, 0, 0, 0};
    __builtin_amdgcn_tensor_load_to_lds(g0, g1, z, z, 0);
  }
#else
  unsigned* d = (unsigned*)dst_lds;
  const unsigned* s = (const unsigned*)src;
  for (unsigned r = 0; r < h; ++r)
    for (unsigned i = threadIdx.x; i < w_dw; i += blockDim.x)
      d[r * w_dw + i] = s[r * stride_dw + i];
#endif
}

__device__ __forceinline__ void stage_commit() {
#if USE_TDM
  if ((threadIdx.x >> 5) == 0) __builtin_amdgcn_s_wait_tensorcnt(0);
#endif
  __syncthreads();
}

// Pack a row-major f32 B matrix [K x N] into WMMA B-fragment tiles:
// packed[((kt*Ntiles+nt)*32+lane)*16+i] = B[kt*32+(i>>3)*16+(lane>>4)*8+(i&7)][nt*16+(lane&15)]
__global__ void pack_b_kernel(const float* __restrict__ B,
                              __bf16* __restrict__ packed, int K, int N) {
  const int idx = blockIdx.x * blockDim.x + threadIdx.x;
  const int total = (K / 32) * (N / 16) * 512;
  if (idx >= total) return;
  const int i    = idx & 15;
  const int lane = (idx >> 4) & 31;
  const int tile = idx >> 9;
  const int ntiles = N / 16;
  const int kt = tile / ntiles;
  const int nt = tile % ntiles;
  const int k = kt * 32 + (i >> 3) * 16 + (lane >> 4) * 8 + (i & 7);
  const int n = nt * 16 + (lane & 15);
  packed[idx] = (__bf16)B[(size_t)k * N + n];
}

__global__ void zero_f32_kernel(float* __restrict__ p, long long n) {
  long long i = (long long)blockIdx.x * blockDim.x + threadIdx.x;
  if (i < n) p[i] = 0.0f;
}

__global__ void copy_f32_kernel(float* __restrict__ dst,
                                const float* __restrict__ src, long long n) {
  long long i = (long long)blockIdx.x * blockDim.x + threadIdx.x;
  if (i < n) dst[i] = src[i];
}

// ---------------------------------------------------------------------------
// Message kernel: msg = selu([h[sf], h[ss]] @ W_msg + b); agg[ss] += msg
// One wave per 16x16 tile, K = 2D = 128 (4 WMMAs). W_msg staged in LDS (16KB).
// ---------------------------------------------------------------------------
__global__ void __launch_bounds__(256) msg_kernel(
    const float* __restrict__ h, const int* __restrict__ sf,
    const int* __restrict__ ss, const __bf16* __restrict__ Wp,
    const float* __restrict__ bias, float* __restrict__ agg) {
  __shared__ __attribute__((aligned(32))) __bf16 sW[16 * 512];  // 16KB
  stage_tiles(sW, Wp, 4096, 1, 4096);
  stage_commit();

  const int lane = threadIdx.x & 31;
  const int wave = blockIdx.x * 8 + (threadIdx.x >> 5);
  const int nt = wave & 3;        // 64/16 col tiles
  const int rt = wave >> 2;
  if (rt >= kM / 16) return;
  const int row0 = rt * 16;
  const int m = lane & 15;
  const int half = lane >> 4;
  const int r = row0 + m;
  const float* rowf = h + (size_t)sf[r] * kD;
  const float* rows = h + (size_t)ss[r] * kD;
  __builtin_prefetch(rows, 0, 3);  // consumed at kt=2,3 -> global_prefetch_b8

  v8f acc = {};
#pragma unroll
  for (int kt = 0; kt < 4; ++kt) {
    const int k0 = kt * 32;
    const float* src = (k0 < kD) ? (rowf + k0) : (rows + (k0 - kD));
    v16bf a = a_frag_f32(src, half);
    v16bf b = b_frag(sW + ((kt * 4 + nt) << 9), lane);
    acc = wmma_bf16(a, b, acc);
  }

  const int n = lane & 15;
  const int col = nt * 16 + n;
  const float bn = bias[col];
#pragma unroll
  for (int j = 0; j < 8; ++j) {
    float v = acc[j] + bn;
    v = (v > 0.0f) ? kSeluScale * v : kSeluScaleAlpha * (__expf(v) - 1.0f);
    const int dest = ss[row0 + j + half * 8];
    atomicAdd(&agg[(size_t)dest * kD + col], v);  // global_atomic_add_f32
  }
}

// ---------------------------------------------------------------------------
// Update GRU: h_new = GRU(agg, h_old), W_upd/U_upd [64 x 192] staged in LDS
// (48KB). One wave owns a 16-row x 16-col gate slab (6 accs, 12 WMMAs).
// ---------------------------------------------------------------------------
__global__ void __launch_bounds__(256) upd_kernel(
    const float* __restrict__ agg, const float* __restrict__ h_old,
    const __bf16* __restrict__ Wp, const __bf16* __restrict__ Up,
    const float* __restrict__ bupd, float* __restrict__ h_new) {
  __shared__ __attribute__((aligned(32))) __bf16 sWU[2 * 24 * 512];  // 48KB
  stage_tiles(sWU, Wp, 6144, 1, 6144);
  stage_tiles(sWU + 24 * 512, Up, 6144, 1, 6144);
  stage_commit();
  const __bf16* sW = sWU;
  const __bf16* sU = sWU + 24 * 512;

  const int lane = threadIdx.x & 31;
  const int wave = blockIdx.x * 8 + (threadIdx.x >> 5);
  const int slab = wave & 3;      // 64/16 slabs
  const int rt = wave >> 2;
  if (rt >= kE / 16) return;
  const int row0 = rt * 16;
  const int c = slab * 16;
  const int m = lane & 15;
  const int half = lane >> 4;

  v8f xz = {}, xr = {}, xh = {}, hz = {}, hr = {}, hh = {};
#pragma unroll
  for (int kt = 0; kt < 2; ++kt) {  // K = 64
    const float* arow = agg   + (size_t)(row0 + m) * kD + kt * 32;
    const float* hrow = h_old + (size_t)(row0 + m) * kD + kt * 32;
    v16bf ax = a_frag_f32(arow, half);
    v16bf ah = a_frag_f32(hrow, half);
    v16bf b;
    b = b_frag(sW + ((kt * 12 + (0 * 4 + slab)) << 9), lane); xz = wmma_bf16(ax, b, xz);
    b = b_frag(sW + ((kt * 12 + (1 * 4 + slab)) << 9), lane); xr = wmma_bf16(ax, b, xr);
    b = b_frag(sW + ((kt * 12 + (2 * 4 + slab)) << 9), lane); xh = wmma_bf16(ax, b, xh);
    b = b_frag(sU + ((kt * 12 + (0 * 4 + slab)) << 9), lane); hz = wmma_bf16(ah, b, hz);
    b = b_frag(sU + ((kt * 12 + (1 * 4 + slab)) << 9), lane); hr = wmma_bf16(ah, b, hr);
    b = b_frag(sU + ((kt * 12 + (2 * 4 + slab)) << 9), lane); hh = wmma_bf16(ah, b, hh);
  }

  const int n = lane & 15;
  const float bxz = bupd[0 * 3 * kD + 0 * kD + c + n];
  const float bxr = bupd[0 * 3 * kD + 1 * kD + c + n];
  const float bxh = bupd[0 * 3 * kD + 2 * kD + c + n];
  const float bhz = bupd[1 * 3 * kD + 0 * kD + c + n];
  const float bhr = bupd[1 * 3 * kD + 1 * kD + c + n];
  const float bhh = bupd[1 * 3 * kD + 2 * kD + c + n];
#pragma unroll
  for (int j = 0; j < 8; ++j) {
    const int row = row0 + j + half * 8;
    const float z    = sigmoidf_((xz[j] + bxz) + (hz[j] + bhz));
    const float rg   = sigmoidf_((xr[j] + bxr) + (hr[j] + bhr));
    const float cand = tanhf((xh[j] + bxh) + rg * (hh[j] + bhh));
    const float ho = h_old[(size_t)row * kD + c + n];
    h_new[(size_t)row * kD + c + n] = z * ho + (1.0f - z) * cand;
  }
}

// ---------------------------------------------------------------------------
// Readout GRU (fused): h_ro_new = GRU_R(x = h_new, h_ro).
// Block = 2 row-tiles x 4 gate slabs so only half of U_ro's column tiles are
// needed -> 48KB LDS (12 rows x 4 tiles, TDM 2D strided copy).
// W_ro (K=64 path) stays in global/WGP$. 18 WMMAs per wave.
// ---------------------------------------------------------------------------
__global__ void __launch_bounds__(256) ro_kernel(
    const float* __restrict__ x, const float* __restrict__ hro_old,
    const __bf16* __restrict__ Wp, const __bf16* __restrict__ Up,
    const float* __restrict__ bro, float* __restrict__ hro_new) {
  __shared__ __attribute__((aligned(32))) __bf16 sU[12 * 4 * 512];  // 48KB
  const int halfsel = blockIdx.x & 1;  // which 4 of the 8 slabs
  // global tile (kt*24 + g*8 + halfsel*4 + s4) == rows r=kt*3+g (stride 8
  // tiles), 4 consecutive tiles per row -> 2D: 12 x 1024 dw, stride 2048 dw.
  stage_tiles(sU, Up + (size_t)halfsel * 4 * 512, 1024, 12, 2048);
  stage_commit();

  const int lane = threadIdx.x & 31;
  const int w = threadIdx.x >> 5;
  const int rt = (blockIdx.x >> 1) * 2 + (w >> 2);
  const int s4 = w & 3;
  const int slab = halfsel * 4 + s4;
  if (rt >= kE / 16) return;
  const int row0 = rt * 16;
  const int c = slab * 16;
  const int m = lane & 15;
  const int half = lane >> 4;

  v8f xz = {}, xr = {}, xh = {}, hz = {}, hr = {}, hh = {};
#pragma unroll
  for (int kt = 0; kt < 2; ++kt) {  // K = 64 (x @ W_ro), B from global
    const float* arow = x + (size_t)(row0 + m) * kD + kt * 32;
    v16bf ax = a_frag_f32(arow, half);
    v16bf b;
    b = b_frag(Wp + ((kt * 24 + (0 * 8 + slab)) << 9), lane); xz = wmma_bf16(ax, b, xz);
    b = b_frag(Wp + ((kt * 24 + (1 * 8 + slab)) << 9), lane); xr = wmma_bf16(ax, b, xr);
    b = b_frag(Wp + ((kt * 24 + (2 * 8 + slab)) << 9), lane); xh = wmma_bf16(ax, b, xh);
  }
#pragma unroll
  for (int kt = 0; kt < 4; ++kt) {  // K = 128 (h_ro @ U_ro), B from LDS
    const float* hrow = hro_old + (size_t)(row0 + m) * kR + kt * 32;
    v16bf ah = a_frag_f32(hrow, half);
    v16bf b;
    b = b_frag(sU + (((kt * 3 + 0) * 4 + s4) << 9), lane); hz = wmma_bf16(ah, b, hz);
    b = b_frag(sU + (((kt * 3 + 1) * 4 + s4) << 9), lane); hr = wmma_bf16(ah, b, hr);
    b = b_frag(sU + (((kt * 3 + 2) * 4 + s4) << 9), lane); hh = wmma_bf16(ah, b, hh);
  }

  const int n = lane & 15;
  const float bxz = bro[0 * 3 * kR + 0 * kR + c + n];
  const float bxr = bro[0 * 3 * kR + 1 * kR + c + n];
  const float bxh = bro[0 * 3 * kR + 2 * kR + c + n];
  const float bhz = bro[1 * 3 * kR + 0 * kR + c + n];
  const float bhr = bro[1 * 3 * kR + 1 * kR + c + n];
  const float bhh = bro[1 * 3 * kR + 2 * kR + c + n];
#pragma unroll
  for (int j = 0; j < 8; ++j) {
    const int row = row0 + j + half * 8;
    const float z    = sigmoidf_((xz[j] + bxz) + (hz[j] + bhz));
    const float rg   = sigmoidf_((xr[j] + bxr) + (hr[j] + bhr));
    const float cand = tanhf((xh[j] + bxh) + rg * (hh[j] + bhh));
    const float ho = hro_old[(size_t)row * kR + c + n];
    hro_new[(size_t)row * kR + c + n] = z * ho + (1.0f - z) * cand;
  }
}

// per_graph[gid[e]] += h_ro[e]  (graph ids sorted -> good atomic locality)
__global__ void graph_scatter_kernel(const float* __restrict__ hro,
                                     const int* __restrict__ gid,
                                     float* __restrict__ pg) {
  long long idx = (long long)blockIdx.x * blockDim.x + threadIdx.x;
  if (idx >= (long long)kE * kR) return;
  const int e = (int)(idx >> 7);
  const int j = (int)(idx & (kR - 1));
  atomicAdd(&pg[(size_t)gid[e] * kR + j], hro[idx]);
}

__global__ void out_kernel(const float* __restrict__ pg,
                           const float* __restrict__ W_out,
                           const float* __restrict__ b_out,
                           float* __restrict__ out) {
  const int g = blockIdx.x * blockDim.x + threadIdx.x;
  if (g >= kG) return;
  float s = b_out[0];
#pragma unroll 8
  for (int j = 0; j < kR; ++j) s += pg[(size_t)g * kR + j] * W_out[j];
  out[g] = s;
}

// ---------------------------------------------------------------------------
extern "C" void kernel_launch(void* const* d_in, const int* in_sizes, int n_in,
                              void* d_out, int out_size, void* d_ws,
                              size_t ws_size, hipStream_t stream) {
  (void)in_sizes; (void)n_in; (void)out_size; (void)ws_size;

  const float* link_state = (const float*)d_in[0];
  const int*   sf         = (const int*)d_in[1];
  const int*   ss         = (const int*)d_in[2];
  const int*   gid        = (const int*)d_in[3];
  // d_in[4] = num_edges scalar (fixed to kE)
  const float* W_msg = (const float*)d_in[5];
  const float* b_msg = (const float*)d_in[6];
  const float* W_upd = (const float*)d_in[7];
  const float* U_upd = (const float*)d_in[8];
  const float* b_upd = (const float*)d_in[9];
  const float* W_ro  = (const float*)d_in[10];
  const float* U_ro  = (const float*)d_in[11];
  const float* b_ro  = (const float*)d_in[12];
  const float* W_out = (const float*)d_in[13];
  const float* b_out = (const float*)d_in[14];
  float* out = (float*)d_out;

  char* ws = (char*)d_ws;
  size_t off = 0;
  auto carve = [&](size_t bytes) -> void* {
    void* p = ws + off;
    off += (bytes + 255) & ~(size_t)255;
    return p;
  };

  __bf16* Wmsg_p = (__bf16*)carve((size_t)2 * kD * kD * 2);
  __bf16* Wupd_p = (__bf16*)carve((size_t)kD * 3 * kD * 2);
  __bf16* Uupd_p = (__bf16*)carve((size_t)kD * 3 * kD * 2);
  __bf16* Wro_p  = (__bf16*)carve((size_t)kD * 3 * kR * 2);
  __bf16* Uro_p  = (__bf16*)carve((size_t)kR * 3 * kR * 2);
  float* hA  = (float*)carve((size_t)kE * kD * 4);
  float* hB  = (float*)carve((size_t)kE * kD * 4);
  float* agg = (float*)carve((size_t)kE * kD * 4);
  float* roA = (float*)carve((size_t)kE * kR * 4);
  float* roB = (float*)carve((size_t)kE * kR * 4);
  float* pg  = (float*)carve((size_t)kG * kR * 4);

  auto pack = [&](const float* src, __bf16* dst, int K, int N) {
    int total = (K / 32) * (N / 16) * 512;
    pack_b_kernel<<<(total + 255) / 256, 256, 0, stream>>>(src, dst, K, N);
  };
  pack(W_msg, Wmsg_p, 2 * kD, kD);
  pack(W_upd, Wupd_p, kD, 3 * kD);
  pack(U_upd, Uupd_p, kD, 3 * kD);
  pack(W_ro,  Wro_p,  kD, 3 * kR);
  pack(U_ro,  Uro_p,  kR, 3 * kR);

  copy_f32_kernel<<<(kE * kD + 255) / 256, 256, 0, stream>>>(
      hA, link_state, (long long)kE * kD);
  zero_f32_kernel<<<(kE * kR + 255) / 256, 256, 0, stream>>>(
      roA, (long long)kE * kR);

  float* h_cur = hA;  float* h_nxt = hB;
  float* ro_cur = roA; float* ro_nxt = roB;

  // msg: (M/16)*4 waves = 100000 -> 12500 blocks of 8 waves
  // upd: (E/16)*4 waves = 25000  -> 3125 blocks
  // ro : (E/16)*8 waves = 50000  -> 6250 blocks (2 row-tiles x 4 slabs each)
  for (int t = 0; t < kT; ++t) {
    zero_f32_kernel<<<(kE * kD + 255) / 256, 256, 0, stream>>>(
        agg, (long long)kE * kD);
    msg_kernel<<<12500, 256, 0, stream>>>(h_cur, sf, ss, Wmsg_p, b_msg, agg);
    upd_kernel<<<3125, 256, 0, stream>>>(agg, h_cur, Wupd_p, Uupd_p, b_upd,
                                         h_nxt);
    ro_kernel<<<6250, 256, 0, stream>>>(h_nxt, ro_cur, Wro_p, Uro_p, b_ro,
                                        ro_nxt);
    float* tmp;
    tmp = h_cur;  h_cur = h_nxt;   h_nxt = tmp;
    tmp = ro_cur; ro_cur = ro_nxt; ro_nxt = tmp;
  }

  zero_f32_kernel<<<(kG * kR + 255) / 256, 256, 0, stream>>>(
      pg, (long long)kG * kR);
  graph_scatter_kernel<<<(int)(((long long)kE * kR + 255) / 256), 256, 0,
                         stream>>>(ro_cur, gid, pg);
  out_kernel<<<(kG + 255) / 256, 256, 0, stream>>>(pg, W_out, b_out, out);
}
